// CombinedLoss_2121713845091
// MI455X (gfx1250) — matrix operand order repaired and checked by reading, not compiled, not run on hardware
//
#include <hip/hip_runtime.h>
#include <hip/hip_bf16.h>

typedef __attribute__((ext_vector_type(16))) _Float16 v16h;
typedef __attribute__((ext_vector_type(8)))  _Float16 v8h;
typedef __attribute__((ext_vector_type(8)))  float    v8f;
typedef __attribute__((ext_vector_type(4)))  unsigned int u32x4;
typedef __attribute__((ext_vector_type(8)))  int          i32x8;
typedef __attribute__((ext_vector_type(4)))  int          i32x4;

// ---------------- accumulator slots (front of d_ws) ----------------
enum { ACC_SL1 = 0, ACC_MSE, ACC_CT0, ACC_CT1, ACC_CP0, ACC_CP1,
       ACC_ST, ACC_SP, ACC_HIST, ACC_PERC, ACC_N };

__global__ void zero_acc_kernel(float* acc) {
  if (threadIdx.x < 64) acc[threadIdx.x] = 0.f;
}

// ------------- pack y_true(2) + y_pred(2) -> [4][256][256][3] f16 NHWC -------------
__global__ __launch_bounds__(256) void prep_input_kernel(
    const float* __restrict__ yt, const float* __restrict__ yp,
    _Float16* __restrict__ out) {
  const int HW = 256 * 256;
  int idx = blockIdx.x * 256 + threadIdx.x;
  const int total = 4 * HW * 3;
  if (idx >= total) return;
  int c = idx % 3;
  int p = (idx / 3) % HW;
  int n = idx / (3 * HW);
  const float* src = (n < 2) ? yt : yp;
  int b = n & 1;
  out[idx] = (_Float16)src[(b * 3 + c) * HW + p];
}

// ------------- OIHW f32 -> [COUT][KPAD] f16, K tap-major (ky,kx,ci), zero-padded ----
template <int CIN, int COUT>
__global__ __launch_bounds__(256) void prep_weight_kernel(
    const float* __restrict__ w, _Float16* __restrict__ out) {
  constexpr int K = CIN * 9;
  constexpr int KPAD = ((K + 31) / 32) * 32;
  int idx = blockIdx.x * 256 + threadIdx.x;
  const int total = COUT * KPAD;
  if (idx >= total) return;
  int k = idx % KPAD;
  int co = idx / KPAD;
  _Float16 v = (_Float16)0.f;
  if (k < K) {
    int tap = k / CIN;
    int ci = k - tap * CIN;
    int ky = tap / 3, kx = tap - ky * 3;
    v = (_Float16)w[((co * CIN + ci) * 3 + ky) * 3 + kx];
  }
  out[idx] = v;
}

// ---- TDM: load 2D weight tile [rows=64][32 halves] (row stride = KPAD halves)
// into LDS with 8-half row padding (pad 4 DWORDs every 16 DWORDs). ----
__device__ __forceinline__ void tdm_load_b_tile(const _Float16* gsrc,
                                                unsigned lds_off,
                                                long long stride0_elems) {
  unsigned long long ga = (unsigned long long)(uintptr_t)gsrc;
  u32x4 g0;
  g0[0] = __builtin_amdgcn_readfirstlane(1u);  // count=1, user mode
  g0[1] = __builtin_amdgcn_readfirstlane(lds_off);
  g0[2] = __builtin_amdgcn_readfirstlane((unsigned)(ga & 0xffffffffull));
  g0[3] = __builtin_amdgcn_readfirstlane(
      (unsigned)((ga >> 32) & 0x01ffffffull) | (2u << 30));  // type=2 (image)
  // group1: data_size=2B(code1), pad_enable, pad_interval=16DW(code3),
  // pad_amount=4DW(code3); tensor_dim0=32, tensor_dim1=64, tile=[32,64]
  i32x8 g1;
  g1[0] = __builtin_amdgcn_readfirstlane(
      (int)((1u << 16) | (1u << 20) | (3u << 22) | (3u << 25)));
  g1[1] = __builtin_amdgcn_readfirstlane((int)(32u << 16));   // dim0 lo16
  g1[2] = __builtin_amdgcn_readfirstlane((int)(64u << 16));   // dim0 hi=0, dim1 lo16=64
  g1[3] = __builtin_amdgcn_readfirstlane((int)(32u << 16));   // dim1 hi=0, tile_dim0=32
  g1[4] = __builtin_amdgcn_readfirstlane((int)64);            // tile_dim1=64, tile_dim2=0
  g1[5] = __builtin_amdgcn_readfirstlane((int)(unsigned)(stride0_elems & 0xffffffffll));
  g1[6] = __builtin_amdgcn_readfirstlane((int)(unsigned)((stride0_elems >> 32) & 0xffffll));
  g1[7] = 0;
  i32x4 z = {0, 0, 0, 0};
#if __clang_major__ >= 23
  i32x8 z8 = {0, 0, 0, 0, 0, 0, 0, 0};
  __builtin_amdgcn_tensor_load_to_lds(g0, g1, z, z, z8, 0);
#else
  __builtin_amdgcn_tensor_load_to_lds(g0, g1, z, z, 0);
#endif
}

// ------------- implicit-GEMM 3x3 SAME conv + bias + ReLU via WMMA f16 -------------
// in  : [4][H][W][CIN] f16 NHWC ; wgt : [COUT][KPAD] f16 ; out : [4][H][W][COUT]
// 256 thr = 8 waves; tile M=128 x N=64; wave w owns rows [w*16,+16), 4 N-frags.
// A LDS rows hold K permuted {0..7,16..23,8..15,24..31} -> lane frag = 2x b128.
// B LDS is [chan][k] (TDM-deposited, padded) -> lane frag = 2x b128.
// Epilogue stages D tile in LDS, then coalesced b128 stores.
template <int CIN, int COUT, int H, int W>
__global__ __launch_bounds__(256) void conv3x3_wmma_kernel(
    const _Float16* __restrict__ in, const _Float16* __restrict__ wgt,
    const float* __restrict__ bias, _Float16* __restrict__ out) {
  constexpr int K = CIN * 9;
  constexpr int KCH = (K + 31) / 32;
  constexpr int KPAD = KCH * 32;

  __shared__ v8h As8[128][5];  // 128 rows x (32 halves + 8 pad)
  __shared__ v8h Bs8[64][5];   // 64 chans x (32 halves + 8 pad)
  __shared__ v8h Ds8[128][8];  // output tile: 128 rows x 64 halves

  const int n = blockIdx.z;
  const int nTilesN = COUT / 64;
  const int tileN = blockIdx.x % nTilesN;
  const int tileM = blockIdx.x / nTilesN;
  const int m0 = tileM * 128;
  const int n0 = tileN * 64;
  const int tid = threadIdx.x;
  const int lane = tid & 31;
  const int wave = tid >> 5;
  const int laneHi = (lane >> 4) & 1;
  const int lane16 = lane & 15;

  // A staging: 2 threads per row, 16 contiguous k each
  const int ar = tid >> 1;
  const int ac0 = (tid & 1) * 16;
  const int am = m0 + ar;
  const int ay = am / W;
  const int ax = am % W;
  const v8h zero8 = {};

  auto loadA = [&](int kc, v8h& r0, v8h& r1) {
    if constexpr (CIN >= 32) {
      const int kg0 = kc * 32;
      const int tap = kg0 / CIN;  // chunk lies inside one tap (CIN % 32 == 0)
      const int ky = tap / 3, kx = tap - ky * 3;
      const int iy = ay + ky - 1, ix = ax + kx - 1;
      const int ci0 = kg0 - tap * CIN + ac0;
      r0 = zero8; r1 = zero8;
      if (iy >= 0 && iy < H && ix >= 0 && ix < W) {
        const v8h* p = (const v8h*)&in[(((size_t)n * H + iy) * W + ix) * CIN + ci0];
        r0 = p[0];
        r1 = p[1];
      }
    } else {  // CIN == 3, single chunk, K=27: scalar gather
      r0 = zero8; r1 = zero8;
      #pragma unroll
      for (int c = 0; c < 16; ++c) {
        int kg = kc * 32 + ac0 + c;
        _Float16 v = (_Float16)0.f;
        if (kg < K) {
          int tap = kg / CIN;
          int ci = kg - tap * CIN;
          int ky = tap / 3, kx = tap - ky * 3;
          int iy = ay + ky - 1, ix = ax + kx - 1;
          if (iy >= 0 && iy < H && ix >= 0 && ix < W)
            v = in[(((size_t)n * H + iy) * W + ix) * CIN + ci];
        }
        if (c < 8) r0[c] = v; else r1[c - 8] = v;
      }
    }
  };

  v8f acc[4] = {};
  v8h a0, a1;
  loadA(0, a0, a1);
  const unsigned bs_lds = (unsigned)(uintptr_t)&Bs8[0][0];

  for (int kc = 0; kc < KCH; ++kc) {
    __syncthreads();  // previous chunk's LDS consumers are done
    if (wave == 0)    // one TDM per block: B tile [64][32] -> Bs8 (padded rows)
      tdm_load_b_tile(wgt + (size_t)n0 * KPAD + kc * 32, bs_lds, KPAD);
    As8[ar][(tid & 1)] = a0;
    As8[ar][(tid & 1) + 2] = a1;
    if (kc + 1 < KCH) loadA(kc + 1, a0, a1);  // overlap next chunk's HBM loads
    if (wave == 0) __builtin_amdgcn_s_wait_tensorcnt(0);
    __syncthreads();

    v8h al = As8[wave * 16 + lane16][laneHi * 2];
    v8h ah = As8[wave * 16 + lane16][laneHi * 2 + 1];
    v16h a = __builtin_shufflevector(al, ah, 0, 1, 2, 3, 4, 5, 6, 7,
                                     8, 9, 10, 11, 12, 13, 14, 15);
    #pragma unroll
    for (int f = 0; f < 4; ++f) {
      v8h bl = Bs8[f * 16 + lane16][laneHi];
      v8h bh = Bs8[f * 16 + lane16][laneHi + 2];
      v16h b = __builtin_shufflevector(bl, bh, 0, 1, 2, 3, 4, 5, 6, 7,
                                       8, 9, 10, 11, 12, 13, 14, 15);
      acc[f] = __builtin_amdgcn_wmma_f32_16x16x32_f16(
          false, a, false, b, (short)0, acc[f], false, false);
    }
  }

  // ---- epilogue: bias+ReLU into LDS tile, then coalesced b128 stores ----
  _Float16* dtile = (_Float16*)Ds8;
  #pragma unroll
  for (int f = 0; f < 4; ++f) {
    int c = f * 16 + lane16;  // column within tile
    float bv = bias[n0 + c];
    #pragma unroll
    for (int r = 0; r < 8; ++r) {
      int m = wave * 16 + r + 8 * laneHi;  // row within tile
      float val = acc[f][r] + bv;
      val = val > 0.f ? val : 0.f;
      dtile[m * 64 + c] = (_Float16)val;
    }
  }
  __syncthreads();
  {
    int m = tid >> 1;           // 2 threads per row
    int s0 = (tid & 1) * 4;     // 4 v8h each
    size_t gbase = ((size_t)n * H * W + m0 + m) * COUT + n0;
    v8h* dst = (v8h*)&out[gbase];
    #pragma unroll
    for (int j = 0; j < 4; ++j) dst[s0 + j] = Ds8[m][s0 + j];
  }
}

// ------------- 2x2 max pool, NHWC f16, 8 channels per thread -------------
template <int C, int H, int W>
__global__ __launch_bounds__(256) void pool2x2_kernel(
    const _Float16* __restrict__ in, _Float16* __restrict__ out) {
  constexpr int HO = H / 2, WO = W / 2, CV = C / 8;
  size_t idx = (size_t)blockIdx.x * 256 + threadIdx.x;
  size_t total = (size_t)4 * HO * WO * CV;
  if (idx >= total) return;
  int cv = (int)(idx % CV);
  int x = (int)((idx / CV) % WO);
  int y = (int)((idx / ((size_t)CV * WO)) % HO);
  int n = (int)(idx / ((size_t)CV * WO * HO));
  size_t base = (((size_t)n * H + 2 * y) * W + 2 * x) * C;
  v8h a = *((const v8h*)&in[base] + cv);
  v8h b = *((const v8h*)&in[base + C] + cv);
  v8h c2 = *((const v8h*)&in[base + (size_t)W * C] + cv);
  v8h d = *((const v8h*)&in[base + (size_t)W * C + C] + cv);
  v8h m;
  #pragma unroll
  for (int e = 0; e < 8; ++e) {
    _Float16 m0 = a[e] > b[e] ? a[e] : b[e];
    _Float16 m1 = c2[e] > d[e] ? c2[e] : d[e];
    m[e] = m0 > m1 ? m0 : m1;
  }
  ((v8h*)out)[idx] = m;
}

// ------------- block reduction (wave32 shuffle + LDS) -------------
__device__ __forceinline__ float block_sum(float v, float* sred) {
  #pragma unroll
  for (int off = 16; off > 0; off >>= 1) v += __shfl_down(v, off, 32);
  int lane = threadIdx.x & 31, w = threadIdx.x >> 5;
  __syncthreads();
  if (lane == 0) sred[w] = v;
  __syncthreads();
  float r = 0.f;
  if (threadIdx.x == 0) {
    int nw = blockDim.x >> 5;
    for (int i = 0; i < nw; ++i) r += sred[i];
  }
  return r;
}

// ------------- smoothL1 + MSE + per-image color sums -------------
__global__ __launch_bounds__(256) void elem_loss_kernel(
    const float* __restrict__ yt, const float* __restrict__ yp,
    float* __restrict__ acc) {
  __shared__ float sred[8];
  int idx = blockIdx.x * 256 + threadIdx.x;  // exact grid
  float t = yt[idx], p = yp[idx];
  float d = p - t;
  float ad = fabsf(d);
  float sl1 = ad < 1.f ? 0.5f * d * d : ad - 0.5f;
  int b = idx / (3 * 256 * 256);  // block-uniform
  float s1 = block_sum(sl1, sred);
  float s2 = block_sum(d * d, sred);
  float s3 = block_sum(t, sred);
  float s4 = block_sum(p, sred);
  if (threadIdx.x == 0) {
    atomicAdd(&acc[ACC_SL1], s1);
    atomicAdd(&acc[ACC_MSE], s2);
    atomicAdd(&acc[ACC_CT0 + b], s3);
    atomicAdd(&acc[ACC_CP0 + b], s4);
  }
}

// ------------- soft-hist pass 1: total kernel mass per tensor -------------
__global__ __launch_bounds__(256) void hist_norm_kernel(
    const float* __restrict__ yt, const float* __restrict__ yp,
    float* __restrict__ acc) {
  __shared__ float sred[8];
  const int NEL = 2 * 3 * 256 * 256;
  int idx = blockIdx.x * 256 + threadIdx.x;  // grid covers 2*NEL exactly
  int which = idx >= NEL;                    // block-uniform
  const float* src = which ? yp : yt;
  float x = src[which ? idx - NEL : idx];
  float s = 0.f;
  const float inv = 1.f / 255.f;
  #pragma unroll 8
  for (int i = 0; i < 256; ++i) {
    float t = (x - i * inv) * 100.f;  // 1/sigma = 100
    s += __expf(-0.5f * t * t);
  }
  float tot = block_sum(s, sred);
  if (threadIdx.x == 0) atomicAdd(&acc[ACC_ST + which], tot);
}

// ------------- soft-hist pass 2: sum |h_t/S_t - h_p/S_p| -------------
__global__ __launch_bounds__(256) void hist_diff_kernel(
    const float* __restrict__ yt, const float* __restrict__ yp,
    const float* __restrict__ accin, float* __restrict__ acc) {
  __shared__ float sred[8];
  const int NPOS = 3 * 256 * 256;
  int pos = blockIdx.x * 256 + threadIdx.x;  // exact grid
  float invSt = 1.f / accin[ACC_ST];
  float invSp = 1.f / accin[ACC_SP];
  float t0 = yt[pos], t1 = yt[NPOS + pos];
  float p0 = yp[pos], p1 = yp[NPOS + pos];
  float s = 0.f;
  const float inv = 1.f / 255.f;
  for (int i = 0; i < 256; ++i) {
    float e = i * inv;
    float a0 = (t0 - e) * 100.f, a1 = (t1 - e) * 100.f;
    float b0 = (p0 - e) * 100.f, b1 = (p1 - e) * 100.f;
    float ht = __expf(-0.5f * a0 * a0) + __expf(-0.5f * a1 * a1);
    float hp = __expf(-0.5f * b0 * b0) + __expf(-0.5f * b1 * b1);
    s += fabsf(ht * invSt - hp * invSp);
  }
  float tot = block_sum(s, sred);
  if (threadIdx.x == 0) atomicAdd(&acc[ACC_HIST], tot);
}

// ------------- perceptual: sum (f_true - f_pred)^2, v8h-wide -------------
__global__ __launch_bounds__(256) void perc_loss_kernel(
    const _Float16* __restrict__ feat, float* __restrict__ acc) {
  __shared__ float sred[8];
  const size_t SV = (size_t)64 * 64 * 256 / 8;  // v8h per image
  size_t idx = (size_t)blockIdx.x * 256 + threadIdx.x;  // exact grid: 2*SV
  size_t n = idx / SV, r = idx % SV;
  v8h ft = ((const v8h*)feat)[n * SV + r];
  v8h fp = ((const v8h*)feat)[(n + 2) * SV + r];
  float s = 0.f;
  #pragma unroll
  for (int e = 0; e < 8; ++e) {
    float d = (float)ft[e] - (float)fp[e];
    s += d * d;
  }
  float tot = block_sum(s, sred);
  if (threadIdx.x == 0) atomicAdd(&acc[ACC_PERC], tot);
}

// ------------- combine everything -------------
__global__ void finalize_kernel(const float* __restrict__ acc,
                                float* __restrict__ out) {
  if (threadIdx.x != 0 || blockIdx.x != 0) return;
  const float Nel = 2.f * 3 * 256 * 256;
  const float Npos = 3.f * 256 * 256;
  float sl1 = acc[ACC_SL1] / Nel;
  float mse = acc[ACC_MSE] / Nel;
  float perc = acc[ACC_PERC] / (2.f * 64 * 64 * 256);
  float hist = acc[ACC_HIST] / (Npos * 256.f);
  float psnr_l = 40.f + 10.f * log10f(mse);
  float color = 0.5f *
      (fabsf(acc[ACC_CT0] - acc[ACC_CP0]) + fabsf(acc[ACC_CT1] - acc[ACC_CP1])) /
      Npos;
  out[0] = 1.0f * sl1 + 0.06f * perc + 0.05f * hist + 0.0083f * psnr_l +
           0.25f * color;
}

// =====================================================================
extern "C" void kernel_launch(void* const* d_in, const int* in_sizes, int n_in,
                              void* d_out, int out_size, void* d_ws,
                              size_t ws_size, hipStream_t stream) {
  const float* yt = (const float*)d_in[0];
  const float* yp = (const float*)d_in[1];
  const float* Wt[7];
  const float* Bb[7];
  for (int i = 0; i < 7; ++i) {
    Wt[i] = (const float*)d_in[2 + 2 * i];
    Bb[i] = (const float*)d_in[3 + 2 * i];
  }
  float* out = (float*)d_out;

  // ws layout: [0,256) acc | [256,~3.5MB) f16 weights [COUT][KPAD] |
  //            [4MiB,36MiB) bufA | [36MiB,68MiB) bufB
  char* base = (char*)d_ws;
  float* acc = (float*)base;
  _Float16* wbuf = (_Float16*)(base + 256);
  // sizes: 64*32, 64*576, 128*576, 128*1152, 256*1152, 256*2304, 256*2304
  const size_t woff[7] = {0, 2048, 38912, 112640, 260096, 555008, 1144832};
  _Float16* bufA = (_Float16*)(base + (4ull << 20));
  _Float16* bufB = (_Float16*)(base + (4ull << 20) + (32ull << 20));

  zero_acc_kernel<<<1, 64, 0, stream>>>(acc);

  prep_input_kernel<<<(4 * 256 * 256 * 3 + 255) / 256, 256, 0, stream>>>(yt, yp, bufA);
  prep_weight_kernel<3, 64><<<(64 * 32) / 256, 256, 0, stream>>>(Wt[0], wbuf + woff[0]);
  prep_weight_kernel<64, 64><<<(64 * 576) / 256, 256, 0, stream>>>(Wt[1], wbuf + woff[1]);
  prep_weight_kernel<64, 128><<<(128 * 576) / 256, 256, 0, stream>>>(Wt[2], wbuf + woff[2]);
  prep_weight_kernel<128, 128><<<(128 * 1152) / 256, 256, 0, stream>>>(Wt[3], wbuf + woff[3]);
  prep_weight_kernel<128, 256><<<(256 * 1152) / 256, 256, 0, stream>>>(Wt[4], wbuf + woff[4]);
  prep_weight_kernel<256, 256><<<(256 * 2304) / 256, 256, 0, stream>>>(Wt[5], wbuf + woff[5]);
  prep_weight_kernel<256, 256><<<(256 * 2304) / 256, 256, 0, stream>>>(Wt[6], wbuf + woff[6]);

  conv3x3_wmma_kernel<3, 64, 256, 256>
      <<<dim3((256 * 256 / 128) * (64 / 64), 1, 4), 256, 0, stream>>>(
          bufA, wbuf + woff[0], Bb[0], bufB);
  conv3x3_wmma_kernel<64, 64, 256, 256>
      <<<dim3((256 * 256 / 128) * (64 / 64), 1, 4), 256, 0, stream>>>(
          bufB, wbuf + woff[1], Bb[1], bufA);
  pool2x2_kernel<64, 256, 256>
      <<<(4 * 128 * 128 * 64 / 8) / 256, 256, 0, stream>>>(bufA, bufB);
  conv3x3_wmma_kernel<64, 128, 128, 128>
      <<<dim3((128 * 128 / 128) * (128 / 64), 1, 4), 256, 0, stream>>>(
          bufB, wbuf + woff[2], Bb[2], bufA);
  conv3x3_wmma_kernel<128, 128, 128, 128>
      <<<dim3((128 * 128 / 128) * (128 / 64), 1, 4), 256, 0, stream>>>(
          bufA, wbuf + woff[3], Bb[3], bufB);
  pool2x2_kernel<128, 128, 128>
      <<<(4 * 64 * 64 * 128 / 8) / 256, 256, 0, stream>>>(bufB, bufA);
  conv3x3_wmma_kernel<128, 256, 64, 64>
      <<<dim3((64 * 64 / 128) * (256 / 64), 1, 4), 256, 0, stream>>>(
          bufA, wbuf + woff[4], Bb[4], bufB);
  conv3x3_wmma_kernel<256, 256, 64, 64>
      <<<dim3((64 * 64 / 128) * (256 / 64), 1, 4), 256, 0, stream>>>(
          bufB, wbuf + woff[5], Bb[5], bufA);
  conv3x3_wmma_kernel<256, 256, 64, 64>
      <<<dim3((64 * 64 / 128) * (256 / 64), 1, 4), 256, 0, stream>>>(
          bufA, wbuf + woff[6], Bb[6], bufB);

  perc_loss_kernel<<<(2 * 64 * 64 * 256 / 8) / 256, 256, 0, stream>>>(bufB, acc);
  elem_loss_kernel<<<(2 * 3 * 256 * 256) / 256, 256, 0, stream>>>(yt, yp, acc);
  hist_norm_kernel<<<(2 * 2 * 3 * 256 * 256) / 256, 256, 0, stream>>>(yt, yp, acc);
  hist_diff_kernel<<<(3 * 256 * 256) / 256, 256, 0, stream>>>(yt, yp, acc, acc);
  finalize_kernel<<<1, 1, 0, stream>>>(acc, out);

  (void)in_sizes; (void)n_in; (void)out_size; (void)ws_size;
}
